// API_RNNAgent_8186207666474
// MI455X (gfx1250) — matrix operand order repaired and checked by reading, not compiled, not run on hardware
//
#include <hip/hip_runtime.h>
#include <hip/hip_bf16.h>

typedef __attribute__((ext_vector_type(16))) _Float16 v16h;
typedef __attribute__((ext_vector_type(8)))  float    v8f;

#define NAG  10
#define NEN  11
#define NALY 9
#define HDIM 64
#define EFD  10
#define NACT 17
#define BSZ  1024
#define NROW (BSZ*NAG)     /* 10240  */
#define EROW (NROW*NEN)    /* 112640 */
#define AROW (NROW*NALY)   /* 92160  */
#define FRAG_H 512         /* halves per packed B fragment: 32 lanes x 16 */

// ---------------------------------------------------------------------------
// WMMA helper (CDNA5, wave32, 16x16x32 f16 -> f32)
// ---------------------------------------------------------------------------
__device__ __forceinline__ v8f wmma16(v16h a, v16h b, v8f c) {
  return __builtin_amdgcn_wmma_f32_16x16x32_f16(
      /*neg_a=*/false, a, /*neg_b=*/false, b,
      /*c_mod=*/(short)0, c, /*reuse_a=*/false, /*reuse_b=*/false);
}

// A fragment: 16x32 f16 tile from LDS, row-major stride ld (halves).
// Lanes 0-15: M=lane, K in {0..7, 16..23}; lanes 16-31: M=lane-16, K in {8..15, 24..31}.
// Per-lane elements are gathered so the compiler merges them into ds_load_b128.
__device__ __forceinline__ v16h load_a_frag(const _Float16* a, int ld) {
  int l = threadIdx.x & 31;
  int m = l & 15;
  int koff = (l >> 4) << 3;
  v16h f;
#pragma unroll
  for (int e = 0; e < 16; ++e) {
    int v = e >> 1;
    int k = ((v & 4) << 2) + koff + ((v & 3) << 1) + (e & 1);
    f[e] = a[m * ld + k];
  }
  return f;
}

// Packed B fragment: one contiguous 32-byte vector load per lane
// (lowered to global_load_b128 x2, no conversion VALU in the hot loop).
__device__ __forceinline__ v16h load_b_packed(const _Float16* base, int fragId) {
  return *(const v16h*)(base + (size_t)fragId * FRAG_H + (threadIdx.x & 31) * 16);
}

// ---------------------------------------------------------------------------
// Prepack: convert f32 weight matrix W (row-major [K][ld], valid K<kvalid,
// N<nvalid) into WMMA B-fragment layout. fragId = nt*kblocks + kb.
// One wave per 32x16 tile.  Element mapping matches v_wmma B operand:
// lanes 0-15: N=lane, K 0..7,16..23; lanes 16-31: K 8..15,24..31 (pairs per VGPR).
// ---------------------------------------------------------------------------
__global__ void __launch_bounds__(128) prepack_b_kernel(
    const float* __restrict__ W, _Float16* __restrict__ dst,
    int ld, int kvalid, int nvalid, int kblocks, int nfrags) {
  int fragId = blockIdx.x * 4 + (threadIdx.x >> 5);
  if (fragId >= nfrags) return;
  int l = threadIdx.x & 31;
  int nt = fragId / kblocks, kb = fragId - nt * kblocks;
  int n = nt * 16 + (l & 15);
  int koff = (l >> 4) << 3;
  _Float16* out = dst + (size_t)fragId * FRAG_H + l * 16;
#pragma unroll
  for (int e = 0; e < 16; ++e) {
    int v = e >> 1;
    int k = kb * 32 + ((v & 4) << 2) + koff + ((v & 3) << 1) + (e & 1);
    float t = (k < kvalid && n < nvalid) ? W[(size_t)k * ld + n] : 0.0f;
    out[e] = (_Float16)t;
  }
}

// ---------------------------------------------------------------------------
// Hypernet trunk with packed weights.
// winP: 4 frags (nt).  wresP: frags [s*8 + nt*2 + kb].
// ---------------------------------------------------------------------------
__device__ __forceinline__ void hyper_trunk(const _Float16* aX,
    const _Float16* winP, const float* bin_,
    const _Float16* wresP, const float* bres,
    _Float16* hStage, float hF[4][8]) {
  int l = threadIdx.x & 31;
  int n = l & 15;
  int moff = (l >> 4) << 3;
  v16h ax = load_a_frag(aX, 32);
#pragma unroll
  for (int nt = 0; nt < 4; ++nt) {
    v8f c = {};
    c = wmma16(ax, load_b_packed(winP, nt), c);
    float b = bin_[nt * 16 + n];
#pragma unroll
    for (int v = 0; v < 8; ++v) {
      float t = c[v] + b;
      hF[nt][v] = t > 0.0f ? t : 0.0f;
    }
  }
#pragma unroll
  for (int nt = 0; nt < 4; ++nt)
#pragma unroll
    for (int v = 0; v < 8; ++v)
      hStage[(v + moff) * HDIM + nt * 16 + n] = (_Float16)hF[nt][v];

#pragma unroll
  for (int s = 0; s < 2; ++s) {
    v16h a0 = load_a_frag(hStage, HDIM);
    v16h a1 = load_a_frag(hStage + 32, HDIM);
    float nf[4][8];
#pragma unroll
    for (int nt = 0; nt < 4; ++nt) {
      v8f c = {};
      c = wmma16(a0, load_b_packed(wresP, s * 8 + nt * 2 + 0), c);
      c = wmma16(a1, load_b_packed(wresP, s * 8 + nt * 2 + 1), c);
      float b = bres[s * HDIM + nt * 16 + n];
#pragma unroll
      for (int v = 0; v < 8; ++v) {
        float t = c[v] + b;
        nf[nt][v] = (t > 0.0f ? t : 0.0f) + hF[nt][v];
      }
    }
#pragma unroll
    for (int nt = 0; nt < 4; ++nt)
#pragma unroll
      for (int v = 0; v < 8; ++v) {
        hF[nt][v] = nf[nt][v];
        hStage[(v + moff) * HDIM + nt * 16 + n] = (_Float16)nf[nt][v];
      }
  }
}

// ---------------------------------------------------------------------------
// K0: emb = own@fc1 + fc1_b + agent_emb[idx] + action_emb[idx]  -> acc[N][64]
// ---------------------------------------------------------------------------
__global__ void __launch_bounds__(128) ownemb_kernel(
    const float* __restrict__ own, const _Float16* __restrict__ fc1P,
    const float* __restrict__ fc1b, const float* __restrict__ agemb,
    const float* __restrict__ acemb, const int* __restrict__ aidx,
    const int* __restrict__ lidx, float* __restrict__ acc) {
  __shared__ _Float16 aStage[4][16 * 32];
  int tid = threadIdx.x, w = tid >> 5, l = tid & 31;
  int rowBase = (blockIdx.x * 4 + w) * 16;
  for (int i = l; i < 16 * 32; i += 32)
    aStage[w][i] = (_Float16)own[rowBase * 32 + i];
  v16h a = load_a_frag(aStage[w], 32);
  int n = l & 15, moff = (l >> 4) << 3;
#pragma unroll
  for (int nt = 0; nt < 4; ++nt) {
    v8f c = {};
    c = wmma16(a, load_b_packed(fc1P, nt), c);
    float b = fc1b[nt * 16 + n];
#pragma unroll
    for (int v = 0; v < 8; ++v) {
      int row = rowBase + v + moff;
      int ai = aidx[row], li = lidx[row];
      acc[row * HDIM + nt * 16 + n] =
          c[v] + b + agemb[ai * HDIM + nt * 16 + n] + acemb[li * HDIM + nt * 16 + n];
    }
  }
}

// ---------------------------------------------------------------------------
// K1/K2: fused hypernet + per-row einsum (never materializes [R,10,64] weights)
// embRows[r][h] = sum_f x[r][f] * (h2[r] @ wout[:, f*64+h] + bout[f*64+h])
// woutP: frags [nt*2 + kb], nt 0..39.
// ---------------------------------------------------------------------------
__global__ void __launch_bounds__(128) hyper_embed_kernel(
    const float* __restrict__ feats,
    const _Float16* __restrict__ winP, const float* __restrict__ bin_,
    const _Float16* __restrict__ wresP, const float* __restrict__ bres,
    const _Float16* __restrict__ woutP, const float* __restrict__ bout,
    float* __restrict__ embRows) {
  __shared__ _Float16 aStage[4][16 * 32];
  __shared__ _Float16 hStage[4][16 * HDIM];
  __shared__ float    xStage[4][16 * 16];
  int tid = threadIdx.x, w = tid >> 5, l = tid & 31;
  int rowBase = (blockIdx.x * 4 + w) * 16;

  for (int i = l; i < 16 * 32; i += 32) {
    int r = i >> 5, k = i & 31;
    float v = (k < EFD) ? feats[(rowBase + r) * EFD + k] : 0.0f;
    aStage[w][i] = (_Float16)v;
    if (k < 16) xStage[w][r * 16 + k] = v;
  }

  float hF[4][8];
  hyper_trunk(aStage[w], winP, bin_, wresP, bres, hStage[w], hF);

  int n = l & 15, moff = (l >> 4) << 3;
  float emb[4][8];
#pragma unroll
  for (int ct = 0; ct < 4; ++ct)
#pragma unroll
    for (int v = 0; v < 8; ++v) emb[ct][v] = 0.0f;

  v16h a0 = load_a_frag(hStage[w], HDIM);
  v16h a1 = load_a_frag(hStage[w] + 32, HDIM);
#pragma unroll 4
  for (int nt = 0; nt < 40; ++nt) {          // 640 output cols = (f, h) pairs
    v8f c = {};
    c = wmma16(a0, load_b_packed(woutP, nt * 2 + 0), c);
    c = wmma16(a1, load_b_packed(woutP, nt * 2 + 1), c);
    int f = nt >> 2, ct = nt & 3;            // col = f*64 + ct*16 + n
    float bo = bout[nt * 16 + n];
#pragma unroll
    for (int v = 0; v < 8; ++v) {
      float xv = xStage[w][(v + moff) * 16 + f];
      emb[ct][v] += xv * (c[v] + bo);
    }
  }
#pragma unroll
  for (int ct = 0; ct < 4; ++ct)
#pragma unroll
    for (int v = 0; v < 8; ++v)
      embRows[(size_t)(rowBase + v + moff) * HDIM + ct * 16 + n] = emb[ct][v];
}

// ---------------------------------------------------------------------------
// K3: deterministic fixed-order pooling: acc += sum_e embEn + sum_a embAl
// ---------------------------------------------------------------------------
__global__ void __launch_bounds__(256) pool_kernel(
    const float* __restrict__ embEn, const float* __restrict__ embAl,
    float* __restrict__ acc) {
  int i = blockIdx.x * 256 + threadIdx.x;
  if (i >= NROW * HDIM) return;
  int nrow = i >> 6, h = i & 63;
  float s = acc[i];
#pragma unroll
  for (int e = 0; e < NEN; ++e) s += embEn[(size_t)(nrow * NEN + e) * HDIM + h];
#pragma unroll
  for (int a = 0; a < NALY; ++a) s += embAl[(size_t)(nrow * NALY + a) * HDIM + h];
  acc[i] = s;
}

// ---------------------------------------------------------------------------
// K4: attack head hypernets (how: 64-wide weights, hob: scalar bias) per enemy
// ---------------------------------------------------------------------------
__global__ void __launch_bounds__(128) attack_head_kernel(
    const float* __restrict__ feats,
    const _Float16* __restrict__ howWinP, const float* __restrict__ how_bin,
    const _Float16* __restrict__ howWresP, const float* __restrict__ how_bres,
    const _Float16* __restrict__ howWoutP, const float* __restrict__ how_bout,
    const _Float16* __restrict__ hobWinP, const float* __restrict__ hob_bin,
    const _Float16* __restrict__ hobWresP, const float* __restrict__ hob_bres,
    const float* __restrict__ hob_wout, const float* __restrict__ hob_bout,
    float* __restrict__ ow, float* __restrict__ ob) {
  __shared__ _Float16 aStage[4][16 * 32];
  __shared__ _Float16 hStage[4][16 * HDIM];
  int tid = threadIdx.x, w = tid >> 5, l = tid & 31;
  int rowBase = (blockIdx.x * 4 + w) * 16;
  for (int i = l; i < 16 * 32; i += 32) {
    int r = i >> 5, k = i & 31;
    aStage[w][i] = (_Float16)((k < EFD) ? feats[(rowBase + r) * EFD + k] : 0.0f);
  }
  int n = l & 15, moff = (l >> 4) << 3;
  float hF[4][8];

  // how: per-enemy 64-wide attack weights
  hyper_trunk(aStage[w], howWinP, how_bin, howWresP, how_bres, hStage[w], hF);
  {
    v16h a0 = load_a_frag(hStage[w], HDIM);
    v16h a1 = load_a_frag(hStage[w] + 32, HDIM);
#pragma unroll
    for (int nt = 0; nt < 4; ++nt) {
      v8f c = {};
      c = wmma16(a0, load_b_packed(howWoutP, nt * 2 + 0), c);
      c = wmma16(a1, load_b_packed(howWoutP, nt * 2 + 1), c);
      float b = how_bout[nt * 16 + n];
#pragma unroll
      for (int v = 0; v < 8; ++v)
        ow[(size_t)(rowBase + v + moff) * HDIM + nt * 16 + n] = c[v] + b;
    }
  }

  // hob: per-enemy scalar attack bias (64->1: cross-lane dot reduction)
  hyper_trunk(aStage[w], hobWinP, hob_bin, hobWresP, hob_bres, hStage[w], hF);
  {
    float p[8];
#pragma unroll
    for (int v = 0; v < 8; ++v) {
      float s = 0.0f;
#pragma unroll
      for (int ct = 0; ct < 4; ++ct) s += hF[ct][v] * hob_wout[ct * 16 + n];
      p[v] = s;
    }
#pragma unroll
    for (int off = 1; off < 16; off <<= 1)
#pragma unroll
      for (int v = 0; v < 8; ++v) p[v] += __shfl_xor(p[v], off, 32);
    if (n == 0) {
#pragma unroll
      for (int v = 0; v < 8; ++v)
        ob[rowBase + v + moff] = p[v] + hob_bout[0];
    }
  }
}

// ---------------------------------------------------------------------------
// K5: GRU cell + q_normal head (all GEMMs as WMMA, gates as VALU)
// wiP/whP: frags [(gate*4 + ct)*2 + kb].  fc2P: frags [kb].
// ---------------------------------------------------------------------------
__global__ void __launch_bounds__(128) gru_kernel(
    const float* __restrict__ acc, const float* __restrict__ hidden,
    const _Float16* __restrict__ wiP, const _Float16* __restrict__ whP,
    const float* __restrict__ bi, const float* __restrict__ bh,
    const _Float16* __restrict__ fc2P, const float* __restrict__ fc2b,
    float* __restrict__ q, float* __restrict__ hhout) {
  __shared__ _Float16 xStage[4][16 * HDIM];
  __shared__ _Float16 hStage[4][16 * HDIM];
  __shared__ float    hF32[4][16 * HDIM];
  __shared__ _Float16 hhStage[4][16 * HDIM];
  int tid = threadIdx.x, w = tid >> 5, l = tid & 31;
  int rowBase = (blockIdx.x * 4 + w) * 16;

  for (int i = l; i < 16 * HDIM; i += 32) {
    int r = i >> 6, k = i & 63;
    float xv = acc[(size_t)(rowBase + r) * HDIM + k];
    xStage[w][i] = (_Float16)(xv > 0.0f ? xv : 0.0f);   // x = relu(emb sum)
    float hv = hidden[(size_t)(rowBase + r) * HDIM + k];
    hStage[w][i] = (_Float16)hv;
    hF32[w][i] = hv;
  }
  v16h ax0 = load_a_frag(xStage[w], HDIM), ax1 = load_a_frag(xStage[w] + 32, HDIM);
  v16h ah0 = load_a_frag(hStage[w], HDIM), ah1 = load_a_frag(hStage[w] + 32, HDIM);
  int n = l & 15, moff = (l >> 4) << 3;

#pragma unroll
  for (int ct = 0; ct < 4; ++ct) {   // 16-col tile of H; gates r/z/n = 0/1/2
    v8f gir = {}, giz = {}, gin = {}, ghr = {}, ghz = {}, ghn = {};
    gir = wmma16(ax0, load_b_packed(wiP, (0 * 4 + ct) * 2 + 0), gir);
    gir = wmma16(ax1, load_b_packed(wiP, (0 * 4 + ct) * 2 + 1), gir);
    giz = wmma16(ax0, load_b_packed(wiP, (1 * 4 + ct) * 2 + 0), giz);
    giz = wmma16(ax1, load_b_packed(wiP, (1 * 4 + ct) * 2 + 1), giz);
    gin = wmma16(ax0, load_b_packed(wiP, (2 * 4 + ct) * 2 + 0), gin);
    gin = wmma16(ax1, load_b_packed(wiP, (2 * 4 + ct) * 2 + 1), gin);
    ghr = wmma16(ah0, load_b_packed(whP, (0 * 4 + ct) * 2 + 0), ghr);
    ghr = wmma16(ah1, load_b_packed(whP, (0 * 4 + ct) * 2 + 1), ghr);
    ghz = wmma16(ah0, load_b_packed(whP, (1 * 4 + ct) * 2 + 0), ghz);
    ghz = wmma16(ah1, load_b_packed(whP, (1 * 4 + ct) * 2 + 1), ghz);
    ghn = wmma16(ah0, load_b_packed(whP, (2 * 4 + ct) * 2 + 0), ghn);
    ghn = wmma16(ah1, load_b_packed(whP, (2 * 4 + ct) * 2 + 1), ghn);
    float bir = bi[ct * 16 + n], biz = bi[64 + ct * 16 + n], bin_ = bi[128 + ct * 16 + n];
    float bhr = bh[ct * 16 + n], bhz = bh[64 + ct * 16 + n], bhn = bh[128 + ct * 16 + n];
#pragma unroll
    for (int v = 0; v < 8; ++v) {
      float r_ = 1.0f / (1.0f + __expf(-((gir[v] + bir) + (ghr[v] + bhr))));
      float z_ = 1.0f / (1.0f + __expf(-((giz[v] + biz) + (ghz[v] + bhz))));
      float nn = tanhf((gin[v] + bin_) + r_ * (ghn[v] + bhn));
      float hprev = hF32[w][(v + moff) * HDIM + ct * 16 + n];
      float hh = (1.0f - z_) * nn + z_ * hprev;
      hhout[(size_t)(rowBase + v + moff) * HDIM + ct * 16 + n] = hh;
      hhStage[w][(v + moff) * HDIM + ct * 16 + n] = (_Float16)hh;
    }
  }

  // q_normal = hh @ fc2 (64x6, N padded to 16) + fc2_b
  v16h aq0 = load_a_frag(hhStage[w], HDIM);
  v16h aq1 = load_a_frag(hhStage[w] + 32, HDIM);
  v8f qn = {};
  qn = wmma16(aq0, load_b_packed(fc2P, 0), qn);
  qn = wmma16(aq1, load_b_packed(fc2P, 1), qn);
  if (n < 6) {
    float b = fc2b[n];
#pragma unroll
    for (int v = 0; v < 8; ++v)
      q[(size_t)(rowBase + v + moff) * NACT + n] = qn[v] + b;
  }
}

// ---------------------------------------------------------------------------
// K6: q_attack[n][e] = hh[n] . ow[n*11+e] + ob[n*11+e]
// ---------------------------------------------------------------------------
__global__ void __launch_bounds__(256) qattack_kernel(
    const float* __restrict__ hh, const float* __restrict__ ow,
    const float* __restrict__ ob, float* __restrict__ q) {
  int i = blockIdx.x * 256 + threadIdx.x;
  if (i >= EROW) return;
  int nrow = i / NEN, e = i - nrow * NEN;
  const float4* h4 = (const float4*)(hh + (size_t)nrow * HDIM);
  const float4* w4 = (const float4*)(ow + (size_t)i * HDIM);
  float s = 0.0f;
#pragma unroll
  for (int k = 0; k < 16; ++k) {
    float4 a = h4[k], b = w4[k];
    s += a.x * b.x + a.y * b.y + a.z * b.z + a.w * b.w;
  }
  q[(size_t)nrow * NACT + 6 + e] = s + ob[i];
}

// ---------------------------------------------------------------------------
extern "C" void kernel_launch(void* const* d_in, const int* in_sizes, int n_in,
                              void* d_out, int out_size, void* d_ws, size_t ws_size,
                              hipStream_t stream) {
  (void)in_sizes; (void)n_in; (void)out_size; (void)ws_size;
  const float* own      = (const float*)d_in[0];
  const float* enfeats  = (const float*)d_in[1];
  const float* alfeats  = (const float*)d_in[2];
  const float* hidden   = (const float*)d_in[3];
  const float* fc1w     = (const float*)d_in[4];
  const float* fc1b     = (const float*)d_in[5];
  const float* agemb    = (const float*)d_in[6];
  const float* acemb    = (const float*)d_in[7];
  const float* hen_win  = (const float*)d_in[8];
  const float* hen_bin  = (const float*)d_in[9];
  const float* hen_wres = (const float*)d_in[10];
  const float* hen_bres = (const float*)d_in[11];
  const float* hen_wout = (const float*)d_in[12];
  const float* hen_bout = (const float*)d_in[13];
  const float* hal_win  = (const float*)d_in[14];
  const float* hal_bin  = (const float*)d_in[15];
  const float* hal_wres = (const float*)d_in[16];
  const float* hal_bres = (const float*)d_in[17];
  const float* hal_wout = (const float*)d_in[18];
  const float* hal_bout = (const float*)d_in[19];
  const float* gru_wi   = (const float*)d_in[20];
  const float* gru_wh   = (const float*)d_in[21];
  const float* gru_bi   = (const float*)d_in[22];
  const float* gru_bh   = (const float*)d_in[23];
  const float* fc2w     = (const float*)d_in[24];
  const float* fc2b     = (const float*)d_in[25];
  const float* how_win  = (const float*)d_in[26];
  const float* how_bin  = (const float*)d_in[27];
  const float* how_wres = (const float*)d_in[28];
  const float* how_bres = (const float*)d_in[29];
  const float* how_wout = (const float*)d_in[30];
  const float* how_bout = (const float*)d_in[31];
  const float* hob_win  = (const float*)d_in[32];
  const float* hob_bin  = (const float*)d_in[33];
  const float* hob_wres = (const float*)d_in[34];
  const float* hob_bres = (const float*)d_in[35];
  const float* hob_wout = (const float*)d_in[36];
  const float* hob_bout = (const float*)d_in[37];
  const int*   aidx     = (const int*)d_in[38];
  const int*   lidx     = (const int*)d_in[39];

  // -------- workspace carve-out: f32 tensors, then packed f16 fragments ----
  float* ws    = (float*)d_ws;
  float* embEn = ws;                                   // [EROW][64]
  float* embAl = embEn + (size_t)EROW * HDIM;          // [AROW][64]
  float* acc   = embAl + (size_t)AROW * HDIM;          // [NROW][64]
  float* ow    = acc   + (size_t)NROW * HDIM;          // [EROW][64]
  float* ob    = ow    + (size_t)EROW * HDIM;          // [EROW]
  _Float16* pk = (_Float16*)(ob + EROW);               // packed B fragments

  _Float16* fc1P     = pk;  pk += (size_t)4  * FRAG_H;
  _Float16* henWinP  = pk;  pk += (size_t)4  * FRAG_H;
  _Float16* henWresP = pk;  pk += (size_t)16 * FRAG_H;
  _Float16* henWoutP = pk;  pk += (size_t)80 * FRAG_H;
  _Float16* halWinP  = pk;  pk += (size_t)4  * FRAG_H;
  _Float16* halWresP = pk;  pk += (size_t)16 * FRAG_H;
  _Float16* halWoutP = pk;  pk += (size_t)80 * FRAG_H;
  _Float16* howWinP  = pk;  pk += (size_t)4  * FRAG_H;
  _Float16* howWresP = pk;  pk += (size_t)16 * FRAG_H;
  _Float16* howWoutP = pk;  pk += (size_t)8  * FRAG_H;
  _Float16* hobWinP  = pk;  pk += (size_t)4  * FRAG_H;
  _Float16* hobWresP = pk;  pk += (size_t)16 * FRAG_H;
  _Float16* wiP      = pk;  pk += (size_t)24 * FRAG_H;
  _Float16* whP      = pk;  pk += (size_t)24 * FRAG_H;
  _Float16* fc2P     = pk;  pk += (size_t)2  * FRAG_H;

  float* q  = (float*)d_out;                           // [BSZ][NAG][17]
  float* hh = q + (size_t)BSZ * NAG * NACT;            // [BSZ][NAG][64]

  // -------- prepack all weights into WMMA B-fragment layout (once) --------
#define PREPACK(W, DST, LD, KV, NV, KB, NF) \
  prepack_b_kernel<<<((NF) + 3) / 4, 128, 0, stream>>>((W), (DST), (LD), (KV), (NV), (KB), (NF))
  PREPACK(fc1w,    fc1P,    HDIM, 32, HDIM, 1, 4);
  PREPACK(hen_win, henWinP, HDIM, EFD, HDIM, 1, 4);
  PREPACK(hen_wres,          henWresP,      HDIM, HDIM, HDIM, 2, 8);
  PREPACK(hen_wres + 4096,   henWresP + 8 * FRAG_H, HDIM, HDIM, HDIM, 2, 8);
  PREPACK(hen_wout, henWoutP, 640, HDIM, 640, 2, 80);
  PREPACK(hal_win, halWinP, HDIM, EFD, HDIM, 1, 4);
  PREPACK(hal_wres,          halWresP,      HDIM, HDIM, HDIM, 2, 8);
  PREPACK(hal_wres + 4096,   halWresP + 8 * FRAG_H, HDIM, HDIM, HDIM, 2, 8);
  PREPACK(hal_wout, halWoutP, 640, HDIM, 640, 2, 80);
  PREPACK(how_win, howWinP, HDIM, EFD, HDIM, 1, 4);
  PREPACK(how_wres,          howWresP,      HDIM, HDIM, HDIM, 2, 8);
  PREPACK(how_wres + 4096,   howWresP + 8 * FRAG_H, HDIM, HDIM, HDIM, 2, 8);
  PREPACK(how_wout, howWoutP, HDIM, HDIM, HDIM, 2, 8);
  PREPACK(hob_win, hobWinP, HDIM, EFD, HDIM, 1, 4);
  PREPACK(hob_wres,          hobWresP,      HDIM, HDIM, HDIM, 2, 8);
  PREPACK(hob_wres + 4096,   hobWresP + 8 * FRAG_H, HDIM, HDIM, HDIM, 2, 8);
  PREPACK(gru_wi, wiP, 192, HDIM, 192, 2, 24);
  PREPACK(gru_wh, whP, 192, HDIM, 192, 2, 24);
  PREPACK(fc2w,   fc2P, 6, HDIM, 6, 2, 2);
#undef PREPACK

  // -------- main pipeline --------
  ownemb_kernel<<<NROW / 64, 128, 0, stream>>>(own, fc1P, fc1b, agemb, acemb,
                                               aidx, lidx, acc);
  hyper_embed_kernel<<<EROW / 64, 128, 0, stream>>>(enfeats, henWinP, hen_bin,
      henWresP, hen_bres, henWoutP, hen_bout, embEn);
  hyper_embed_kernel<<<AROW / 64, 128, 0, stream>>>(alfeats, halWinP, hal_bin,
      halWresP, hal_bres, halWoutP, hal_bout, embAl);
  pool_kernel<<<(NROW * HDIM + 255) / 256, 256, 0, stream>>>(embEn, embAl, acc);
  attack_head_kernel<<<EROW / 64, 128, 0, stream>>>(enfeats,
      howWinP, how_bin, howWresP, how_bres, howWoutP, how_bout,
      hobWinP, hob_bin, hobWresP, hob_bres, hob_wout, hob_bout, ow, ob);
  gru_kernel<<<NROW / 64, 128, 0, stream>>>(acc, hidden, wiP, whP,
      gru_bi, gru_bh, fc2P, fc2b, q, hh);
  qattack_kernel<<<(EROW + 255) / 256, 256, 0, stream>>>(hh, ow, ob, q);
}
